// DeepLSTM_77060303225285
// MI455X (gfx1250) — matrix operand order repaired and checked by reading, compiled, not run on hardware
//
#include <hip/hip_runtime.h>
#include <stdint.h>

// ---------------------------------------------------------------------------
// DeepLSTM for MI455X (gfx1250): weight-stationary wavefront pipeline.
//  - 8 persistent workgroups (one per layer), fp16 weights resident in LDS
//  - v_wmma_f32_16x16x32_f16 for all GEMMs, f32 accumulation, fp32 cell state
//  - async Global<->LDS (ASYNCcnt) for activation traffic, dev-scope
//  - LDS rows padded (136 f16) so A-fragment ds_load_b128 is bank-conflict-free
//  - lock-free producer/consumer flags between layers (acquire/release, agent)
// ---------------------------------------------------------------------------

#define BATCH 64
#define TT    512
#define LAY   8
#define HID   128
#define G4    512          // 4*H
#define OUTC  65
#define DEPTH 4            // inter-layer ring depth (power of two)
#define XROW  136          // padded LDS row pitch in f16 (272B = 68 banks)

// d_ws byte offsets
#define FLAGS_OFF   0ull           // padded int flags (memset each call)
#define BIAS_OFF    4096ull        // L*512 f32 combined bias
#define WOUT_OFF    20480ull       // 5*4*32*16 f16 w_out fragments (padded to 80 cols)
#define RING_OFF    40960ull       // 7 * DEPTH * 64*128 f16
#define WFRAG_OFF   499712ull      // 8 * 2 * 65536 f16 weight fragments
#define XE_OFF      2596864ull     // T * 64*128 f16 embedded input
#define HLAST_OFF   10985472ull    // T * 64*128 f16 layer-7 output

// LDS byte offsets (dynamic shared, base 0)
#define LDS_WIH   0u
#define LDS_WHH   131072u
#define LDS_X     262144u
#define LDS_H     279552u          // 262144 + 64*272
#define LDS_TOTAL 296960u          // 279552 + 64*272

typedef _Float16 v16h __attribute__((ext_vector_type(16)));
typedef _Float16 v8h  __attribute__((ext_vector_type(8)));
typedef float    v8f  __attribute__((ext_vector_type(8)));

__device__ __forceinline__ float sigf(float x) {
    return 1.0f / (1.0f + __expf(-x));
}
__device__ __forceinline__ float tanhf_fast(float x) {
    float e = __expf(2.0f * x);
    return (e - 1.0f) / (e + 1.0f);
}

// 16x32 f16 A-fragment (ISA 7.12.2): lane<16 -> row=lane, K {0..7,16..23};
// lane>=16 -> row=lane-16, K {8..15,24..31}.
__device__ __forceinline__ v16h load_a_frag(const _Float16* rowp, int kstart) {
    v8h lo = *(const v8h*)(rowp + kstart);
    v8h hi = *(const v8h*)(rowp + kstart + 16);
    v16h a;
#pragma unroll
    for (int i = 0; i < 8; ++i) { a[i] = lo[i]; a[i + 8] = hi[i]; }
    return a;
}

// Async copy: 16B global -> LDS and LDS -> global (INST_OFFSET applies to both
// sides per ISA 08_async_tensor 4.4), device scope so cross-WGP传 data is coherent.
__device__ __forceinline__ void async_g2l_32B(unsigned ldsoff, const void* gsrc) {
    asm volatile("global_load_async_to_lds_b128 %0, %1, off scope:SCOPE_DEV"
                 :: "v"(ldsoff), "v"(gsrc) : "memory");
    asm volatile("global_load_async_to_lds_b128 %0, %1, off offset:16 scope:SCOPE_DEV"
                 :: "v"(ldsoff), "v"(gsrc) : "memory");
}
__device__ __forceinline__ void async_l2g_32B(void* gdst, unsigned ldsoff) {
    asm volatile("global_store_async_from_lds_b128 %0, %1, off scope:SCOPE_DEV"
                 :: "v"(gdst), "v"(ldsoff) : "memory");
    asm volatile("global_store_async_from_lds_b128 %0, %1, off offset:16 scope:SCOPE_DEV"
                 :: "v"(gdst), "v"(ldsoff) : "memory");
}
__device__ __forceinline__ void wait_async0() {
    asm volatile("s_wait_asynccnt 0x0" ::: "memory");
}

// ---------------------------------------------------------------------------
// Prep kernels
// ---------------------------------------------------------------------------

// xe[t][b][h] = (f16) embed[x[b][t]][h]
__global__ void prep_embed_kernel(const int* __restrict__ x,
                                  const float* __restrict__ embed,
                                  _Float16* __restrict__ xe) {
    int idx = blockIdx.x * blockDim.x + threadIdx.x;   // T*B*H = 4194304
    if (idx >= TT * BATCH * HID) return;
    int t   = idx >> 13;
    int rem = idx & 8191;
    int b   = rem >> 7;
    int h   = rem & 127;
    int tok = x[b * TT + t];
    xe[idx] = (_Float16)embed[tok * HID + h];
}

// Weight fragments in WMMA-B order (f16):
// frag[(((nt*4)+kk)*32+lane)*16 + e] = w[l][K][N],
//   K = kk*32 + (lane<16?0:8) + e + (e&8),  N = nt*16 + (lane&15)
__global__ void prep_wfrag_kernel(const float* __restrict__ w_ih,
                                  const float* __restrict__ w_hh,
                                  _Float16* __restrict__ wfrag) {
    int idx = blockIdx.x * blockDim.x + threadIdx.x;   // 8*2*65536 = 1048576
    if (idx >= LAY * 2 * 65536) return;
    int lm  = idx >> 16;
    int l   = lm >> 1;
    int sel = lm & 1;
    int r    = idx & 65535;
    int e    = r & 15;
    int lane = (r >> 4) & 31;
    int kk   = (r >> 9) & 3;
    int nt   = r >> 11;
    int K = kk * 32 + ((lane < 16) ? 0 : 8) + e + (e & 8);
    int N = nt * 16 + (lane & 15);
    const float* src = sel ? w_hh : w_ih;
    wfrag[idx] = (_Float16)src[(l * HID + K) * G4 + N];
}

// combined bias + w_out fragments (padded N to 80, zeros past 65)
__global__ void prep_misc_kernel(const float* __restrict__ b_ih,
                                 const float* __restrict__ b_hh,
                                 const float* __restrict__ w_out,
                                 float* __restrict__ bias,
                                 _Float16* __restrict__ wofrag) {
    int i = blockIdx.x * blockDim.x + threadIdx.x;   // 56*256 = 14336
    if (i < LAY * G4) {
        bias[i] = b_ih[i] + b_hh[i];
        return;
    }
    int j = i - LAY * G4;
    if (j >= 5 * 4 * 32 * 16) return;
    int e    = j & 15;
    int lane = (j >> 4) & 31;
    int kk   = (j >> 9) & 3;
    int nt   = j >> 11;
    int K = kk * 32 + ((lane < 16) ? 0 : 8) + e + (e & 8);
    int N = nt * 16 + (lane & 15);
    wofrag[j] = (_Float16)((N < OUTC) ? w_out[K * OUTC + N] : 0.0f);
}

// ---------------------------------------------------------------------------
// Persistent pipeline kernel: one workgroup per layer, 512 threads (16 waves).
// Dynamic LDS: [wih 128KB][whh 128KB][x_buf 17KB pad][h_buf 17KB pad] = 290KB
// ---------------------------------------------------------------------------
__global__ __launch_bounds__(512)
void lstm_pipeline_kernel(const _Float16* __restrict__ xe,
                          const _Float16* __restrict__ wfrag,
                          const float* __restrict__ bias,
                          _Float16* __restrict__ ring,
                          _Float16* __restrict__ hlast,
                          int* __restrict__ flags,
                          float* __restrict__ outp) {
    extern __shared__ char smem[];
    _Float16* wih = (_Float16*)(smem + LDS_WIH);
    _Float16* whh = (_Float16*)(smem + LDS_WHH);
    _Float16* xsb = (_Float16*)(smem + LDS_X);     // 64 rows, pitch XROW
    _Float16* hsb = (_Float16*)(smem + LDS_H);     // 64 rows, pitch XROW

    const int l      = blockIdx.x;
    const int tid    = threadIdx.x;
    const int lane   = tid & 31;
    const int wave   = tid >> 5;
    const int lanelo = lane & 15;
    const int hibit  = (lane >> 4) & 1;
    const int kbase0 = hibit ? 8 : 0;

    // chunk mapping for 16KB activation transfers: tid -> (row, 32B chunk)
    const int crow = tid >> 3;
    const int cq8  = tid & 7;
    const unsigned ldsx_off = (unsigned)(size_t)(xsb + crow * XROW + cq8 * 16);
    const unsigned ldsh_off = (unsigned)(size_t)(hsb + crow * XROW + cq8 * 16);

    // Stage this layer's fp16 weight fragments into LDS via async DMA (once).
    {
        const char* wsrc = (const char*)(wfrag + (size_t)l * 131072);
#pragma unroll
        for (int i = 0; i < 32; ++i) {
            int c = tid + i * 512;                    // 16B chunk id, 16384 total
            unsigned loff = (unsigned)(size_t)(smem + c * 16);
            asm volatile("global_load_async_to_lds_b128 %0, %1, off"
                         :: "v"(loff), "v"(wsrc + (size_t)c * 16) : "memory");
        }
    }
    // Zero h_buf (h_{t=-1} = 0), padded pitch.
    {
        uint32_t* hz = (uint32_t*)(hsb + crow * XROW) + cq8 * 8;
#pragma unroll
        for (int i = 0; i < 8; ++i) hz[i] = 0u;
    }
    wait_async0();
    __syncthreads();

    const float* biasL = bias + l * G4;
    float creg[2][8];
#pragma unroll
    for (int g = 0; g < 2; ++g)
#pragma unroll
        for (int e = 0; e < 8; ++e) creg[g][e] = 0.0f;

    for (int t = 0; t < TT; ++t) {
        // ---- inter-layer sync (thread0 spins, then workgroup barrier) ----
        if (tid == 0) {
            if (l > 0) {
                while (__hip_atomic_load(&flags[(l - 1) * 32], __ATOMIC_ACQUIRE,
                                         __HIP_MEMORY_SCOPE_AGENT) < t + 1)
                    __builtin_amdgcn_s_sleep(1);
            }
            if (l < LAY - 1 && t >= DEPTH) {
                while (__hip_atomic_load(&flags[(l + 1) * 32], __ATOMIC_ACQUIRE,
                                         __HIP_MEMORY_SCOPE_AGENT) < t - DEPTH + 1)
                    __builtin_amdgcn_s_sleep(1);
            }
        }
        __syncthreads();

        // ---- async-stage x_t (dense global rows -> padded LDS rows) ----
        {
            const _Float16* srcb = (l == 0)
                ? xe + (size_t)t * 8192
                : ring + (size_t)((l - 1) * DEPTH + (t & (DEPTH - 1))) * 8192;
            async_g2l_32B(ldsx_off, srcb + tid * 16);
        }
        wait_async0();
        __syncthreads();

        // ---- gates = x @ Wih + h @ Whh + bias, via WMMA f32_16x16x32_f16 ----
        v8f acc[2][4];
#pragma unroll
        for (int grp = 0; grp < 2; ++grp) {
            const int g = grp * 16 + wave;
            const int m = g >> 3;
            const int j = g & 7;
#pragma unroll
            for (int q = 0; q < 4; ++q) {
                float bv = biasL[(j + 8 * q) * 16 + lanelo];
#pragma unroll
                for (int e = 0; e < 8; ++e) acc[grp][q][e] = bv;
            }
            const _Float16* xrow = xsb + (m * 16 + lanelo) * XROW;
            const _Float16* hrow = hsb + (m * 16 + lanelo) * XROW;
#pragma unroll
            for (int kk = 0; kk < 4; ++kk) {
                v16h a = load_a_frag(xrow, kk * 32 + kbase0);
#pragma unroll
                for (int q = 0; q < 4; ++q) {
                    v16h b = *(const v16h*)(wih + (size_t)(((j + 8 * q) * 4 + kk) * 32 + lane) * 16);
                    acc[grp][q] = __builtin_amdgcn_wmma_f32_16x16x32_f16(
                        false, a, false, b, (short)0, acc[grp][q], false, false);
                }
            }
#pragma unroll
            for (int kk = 0; kk < 4; ++kk) {
                v16h a = load_a_frag(hrow, kk * 32 + kbase0);
#pragma unroll
                for (int q = 0; q < 4; ++q) {
                    v16h b = *(const v16h*)(whh + (size_t)(((j + 8 * q) * 4 + kk) * 32 + lane) * 16);
                    acc[grp][q] = __builtin_amdgcn_wmma_f32_16x16x32_f16(
                        false, a, false, b, (short)0, acc[grp][q], false, false);
                }
            }
        }
        __syncthreads();   // all h_buf reads done before we overwrite it

        // ---- elementwise gate math, c/h update (registers + LDS only) ----
#pragma unroll
        for (int grp = 0; grp < 2; ++grp) {
            const int g = grp * 16 + wave;
            const int m = g >> 3;
            const int j = g & 7;
            const int colh = j * 16 + lanelo;
#pragma unroll
            for (int e = 0; e < 8; ++e) {
                float iv = sigf(acc[grp][0][e]);
                float fv = sigf(acc[grp][1][e]);
                float gv = tanhf_fast(acc[grp][2][e]);
                float ov = sigf(acc[grp][3][e]);
                float c  = fv * creg[grp][e] + iv * gv;
                creg[grp][e] = c;
                float h  = ov * tanhf_fast(c);
                int bb   = m * 16 + e + hibit * 8;     // batch row
                hsb[bb * XROW + colh] = (_Float16)h;   // own recurrence + DMA source
                if (t == TT - 1) {
                    size_t base = (size_t)BATCH * TT * OUTC;
                    outp[base + (size_t)l * 8192 + bb * HID + colh] = h;          // hT
                    outp[base + 65536 + (size_t)l * 8192 + bb * HID + colh] = c;  // cT
                }
            }
        }
        __syncthreads();   // h_buf fully written before async store reads it

        // ---- async-publish h (padded LDS rows -> dense global rows) ----
        {
            _Float16* dstb = (l < LAY - 1)
                ? ring + (size_t)(l * DEPTH + (t & (DEPTH - 1))) * 8192
                : hlast + (size_t)t * 8192;
            async_l2g_32B(dstb + tid * 16, ldsh_off);
        }
        wait_async0();     // each wave drains its own stores
        __syncthreads();   // => all waves' stores complete before release
        if (tid == 0) {
            __hip_atomic_store(&flags[l * 32], t + 1, __ATOMIC_RELEASE,
                               __HIP_MEMORY_SCOPE_AGENT);
        }
    }
}

// ---------------------------------------------------------------------------
// Projection: out[b][t][o] = hlast[t][b][:] @ w_out + b_out, WMMA over 80 cols
// ---------------------------------------------------------------------------
__global__ __launch_bounds__(256)
void proj_kernel(const _Float16* __restrict__ hlast,
                 const _Float16* __restrict__ wofrag,
                 const float* __restrict__ b_out,
                 float* __restrict__ outp) {
    const int lane   = threadIdx.x & 31;
    const int wave   = threadIdx.x >> 5;
    const int lanelo = lane & 15;
    const int hibit  = (lane >> 4) & 1;
    const int kbase0 = hibit ? 8 : 0;
    const int mtile  = blockIdx.x * 8 + wave;          // 0..2047

    v8f acc[5];
#pragma unroll
    for (int nt = 0; nt < 5; ++nt) {
        int col = nt * 16 + lanelo;
        float bv = (col < OUTC) ? b_out[col] : 0.0f;
#pragma unroll
        for (int e = 0; e < 8; ++e) acc[nt][e] = bv;
    }
    const _Float16* arow = hlast + (size_t)(mtile * 16 + lanelo) * HID;
#pragma unroll
    for (int kk = 0; kk < 4; ++kk) {
        v16h a = load_a_frag(arow, kk * 32 + kbase0);
#pragma unroll
        for (int nt = 0; nt < 5; ++nt) {
            v16h b = *(const v16h*)(wofrag + (size_t)((nt * 4 + kk) * 32 + lane) * 16);
            acc[nt] = __builtin_amdgcn_wmma_f32_16x16x32_f16(
                false, a, false, b, (short)0, acc[nt], false, false);
        }
    }
#pragma unroll
    for (int nt = 0; nt < 5; ++nt) {
        int col = nt * 16 + lanelo;
        if (col < OUTC) {
#pragma unroll
            for (int e = 0; e < 8; ++e) {
                int r = mtile * 16 + e + hibit * 8;    // r = t*64 + b
                int t = r >> 6;
                int b = r & 63;
                outp[((size_t)b * TT + t) * OUTC + col] = acc[nt][e];
            }
        }
    }
}

// ---------------------------------------------------------------------------
extern "C" void kernel_launch(void* const* d_in, const int* in_sizes, int n_in,
                              void* d_out, int out_size, void* d_ws, size_t ws_size,
                              hipStream_t stream) {
    (void)in_sizes; (void)n_in; (void)out_size; (void)ws_size;
    const int*   x      = (const int*)d_in[0];
    const float* embed  = (const float*)d_in[1];
    const float* w_ih   = (const float*)d_in[2];
    const float* b_ih   = (const float*)d_in[3];
    const float* w_hh   = (const float*)d_in[4];
    const float* b_hh   = (const float*)d_in[5];
    const float* w_out  = (const float*)d_in[6];
    const float* b_out  = (const float*)d_in[7];
    float* outp = (float*)d_out;

    char* ws = (char*)d_ws;
    int*      flags  = (int*)(ws + FLAGS_OFF);
    float*    bias   = (float*)(ws + BIAS_OFF);
    _Float16* wofrag = (_Float16*)(ws + WOUT_OFF);
    _Float16* ring   = (_Float16*)(ws + RING_OFF);
    _Float16* wfrag  = (_Float16*)(ws + WFRAG_OFF);
    _Float16* xe     = (_Float16*)(ws + XE_OFF);
    _Float16* hlast  = (_Float16*)(ws + HLAST_OFF);

    (void)hipFuncSetAttribute((const void*)lstm_pipeline_kernel,
                              hipFuncAttributeMaxDynamicSharedMemorySize, LDS_TOTAL);

    (void)hipMemsetAsync(ws + FLAGS_OFF, 0, 4096, stream);

    prep_embed_kernel<<<16384, 256, 0, stream>>>(x, embed, xe);
    prep_wfrag_kernel<<<4096, 256, 0, stream>>>(w_ih, w_hh, wfrag);
    prep_misc_kernel<<<56, 256, 0, stream>>>(b_ih, b_hh, w_out, bias, wofrag);

    lstm_pipeline_kernel<<<LAY, 512, LDS_TOTAL, stream>>>(xe, wfrag, bias, ring,
                                                          hlast, flags, outp);
    proj_kernel<<<256, 256, 0, stream>>>(hlast, wofrag, b_out, outp);
}
